// TTTLayer_61538291417805
// MI455X (gfx1250) — compile-verified, matrix operand order checked by reading, not run on hardware
//
#include <hip/hip_runtime.h>
#include <math.h>

// ---------------------------------------------------------------------------
// TTT layer for MI455X (gfx1250), dual-form chunked reformulation.
//   err = (I + strict_lower(A))^{-1} u,  A[t,s] = eta*(x_t.x_s + 1)
//   u = x@(W_init - Wt) + (b_init - bt),  inner = err + (x@Wt + bt)
//   ttt = silu(inner)@Wo + bo ; g = sigmoid([x,ttt]@Wg + bg)
//   out = layernorm(g*ttt + (1-g)*x)
// GEMMs: bf16 operands, fp32 accumulate via v_wmma_f32_16x16x32_bf16.
// LDS staging: double-buffered, A tiles via GLOBAL_LOAD_ASYNC_TO_LDS (ASYNCcnt)
// when the builtin exists, B tiles software-pipelined through VGPRs.
// ---------------------------------------------------------------------------

typedef __attribute__((ext_vector_type(16))) __bf16 v16bf;
typedef __attribute__((ext_vector_type(8)))  float  v8f;
typedef __attribute__((ext_vector_type(4)))  int    v4i;

#if defined(__AMDGCN__) && __has_builtin(__builtin_amdgcn_global_load_async_to_lds_b128)
#define ASYNC_LDS 1
#endif

#ifdef ASYNC_LDS
typedef __attribute__((address_space(1))) v4i as1_v4i;
typedef __attribute__((address_space(3))) v4i as3_v4i;
#endif

#define B_  64
#define S_  512
#define H_  1024
#define I_  1024
#define ETA (0.01f / 1024.0f)

#define BM 128
#define BN 128
#define BK 32
#define LDP 40   // padded LDS row: 80B -> 16B-aligned rows, stride 20 dwords
                 // (20*m mod 64 distinct for m=0..15 -> conflict-free frags)

// Copy 32 contiguous bytes (16 bf16) global -> LDS. Async DMA if available.
__device__ inline void copy32_g2l(const __bf16* __restrict__ gsrc,
                                  __bf16* __restrict__ ldst) {
#ifdef ASYNC_LDS
  as1_v4i* g = (as1_v4i*)(void*)const_cast<__bf16*>(gsrc);
  as3_v4i* l = (as3_v4i*)(void*)ldst;
  __builtin_amdgcn_global_load_async_to_lds_b128(g, l, 0, 0);
  __builtin_amdgcn_global_load_async_to_lds_b128(g + 1, l + 1, 0, 0);
#else
#pragma unroll
  for (int j = 0; j < 16; ++j) ldst[j] = gsrc[j];
#endif
}

__device__ inline void wait_async0() {
#ifdef ASYNC_LDS
#if __has_builtin(__builtin_amdgcn_s_wait_asynccnt)
  __builtin_amdgcn_s_wait_asynccnt(0);
#else
  asm volatile("s_wait_asynccnt 0x0" ::: "memory");
#endif
#endif
}

// Build a 16x32 bf16 WMMA fragment (A layout, ISA 7.12.2) from a padded LDS
// row for matrix row m (= lane&15). lh = lane>>4 selects the K half.
// VGPR v holds K = {base, base+1}, base = ((v&3) + lh*4 + (v>>2)*8)*2.
__device__ inline v16bf frag_from_row(const __bf16* row, int lh) {
  v16bf f;
#pragma unroll
  for (int v = 0; v < 8; ++v) {
    const int kw = (((v & 3) + lh * 4 + (v >> 2) * 8) << 1);
    f[2 * v]     = row[kw];
    f[2 * v + 1] = row[kw + 1];
  }
  return f;
}

__device__ inline v16bf frag_from_row_f32(const float* row, int lh) {
  v16bf f;
#pragma unroll
  for (int v = 0; v < 8; ++v) {
    const int kw = (((v & 3) + lh * 4 + (v >> 2) * 8) << 1);
    f[2 * v]     = (__bf16)row[kw];
    f[2 * v + 1] = (__bf16)row[kw + 1];
  }
  return f;
}

// ---------------------------------------------------------------------------
// Pack: bf16 copies of x and the (combined) weights.
// ---------------------------------------------------------------------------
__global__ void pack_kernel(const float* __restrict__ x,
                            const float* __restrict__ W_init,
                            const float* __restrict__ b_init,
                            const float* __restrict__ Wt,
                            const float* __restrict__ bt,
                            const float* __restrict__ Wo,
                            const float* __restrict__ Wg,
                            __bf16* __restrict__ Xb,
                            __bf16* __restrict__ Wd,
                            __bf16* __restrict__ Wtb,
                            __bf16* __restrict__ Wob,
                            __bf16* __restrict__ Wgx,
                            __bf16* __restrict__ Wgt,
                            float* __restrict__ bias_u)
{
  const size_t n = (size_t)B_ * S_ * H_;
  for (size_t i = (size_t)blockIdx.x * blockDim.x + threadIdx.x; i < n;
       i += (size_t)gridDim.x * blockDim.x) {
    Xb[i] = (__bf16)x[i];
    if (i < (size_t)H_ * I_) {
      Wd[i]  = (__bf16)(W_init[i] - Wt[i]);
      Wtb[i] = (__bf16)Wt[i];
      Wob[i] = (__bf16)Wo[i];
      const size_t h = i / H_, j = i % H_;
      Wgx[i] = (__bf16)Wg[i];                      // Wg[0:H, :]
      Wgt[i] = (__bf16)Wg[(h + H_) * H_ + j];      // Wg[H:2H, :]
      if (i < I_) bias_u[i] = b_init[i] - bt[i];
    }
  }
}

// ---------------------------------------------------------------------------
// Generic GEMM: C[m,n] = sum_k A[m,k]*B[k,n] + bias[n]
// A: [M,K] bf16 row-major, B: [K,N] bf16 row-major, OUT = float or __bf16.
// 256 threads = 8 waves; 128x128x32 tile; wave: 32x64 (2x4 WMMA tiles).
// Double-buffered LDS; A tile staged with async-to-LDS DMA, B tile
// software-pipelined through VGPRs (transposed ds stores).
// ---------------------------------------------------------------------------
template <typename OUT>
__global__ __launch_bounds__(256) void gemm_bf16_kernel(
    const __bf16* __restrict__ A, const __bf16* __restrict__ Bm,
    const float* __restrict__ bias, OUT* __restrict__ C,
    int M, int N, int K)
{
  __shared__ __align__(16) __bf16 As[2][BM][LDP];
  __shared__ __align__(16) __bf16 Bs[2][BN][LDP];   // transposed: Bs[n][k]
  const int tid = threadIdx.x;
  const int m0 = blockIdx.y * BM, n0 = blockIdx.x * BN;
  const int w = tid >> 5, lane = tid & 31;
  const int wr = w >> 1, wc = w & 1;
  const int lm = lane & 15, lh = lane >> 4;

  v8f acc[2][4];
  const v8f vzero = {0.f, 0.f, 0.f, 0.f, 0.f, 0.f, 0.f, 0.f};
#pragma unroll
  for (int i = 0; i < 2; ++i)
#pragma unroll
    for (int j = 0; j < 4; ++j) acc[i][j] = vzero;

  const int ar = tid >> 1, ac = (tid & 1) << 4;     // A: 128 rows x 32
  const int bk = tid >> 3, bn = (tid & 7) << 4;     // B: 32 rows x 128

  __bf16 breg[16];
  // prologue: stage tile 0
  copy32_g2l(A + (size_t)(m0 + ar) * K + ac, &As[0][ar][ac]);
#pragma unroll
  for (int j = 0; j < 16; ++j) breg[j] = Bm[(size_t)bk * N + n0 + bn + j];
#pragma unroll
  for (int j = 0; j < 16; ++j) Bs[0][bn + j][bk] = breg[j];
  wait_async0();
  __syncthreads();

  int buf = 0;
  for (int k0 = 0; k0 < K; k0 += BK, buf ^= 1) {
    const int nbuf = buf ^ 1;
    const bool more = (k0 + BK) < K;
    if (more) {
      // kick off next A tile DMA + next B tile global loads (overlap WMMA)
      copy32_g2l(A + (size_t)(m0 + ar) * K + k0 + BK + ac, &As[nbuf][ar][ac]);
      const __bf16* bsrc = Bm + (size_t)(k0 + BK + bk) * N + n0 + bn;
#pragma unroll
      for (int j = 0; j < 16; ++j) breg[j] = bsrc[j];
    }
    v16bf af[2], bfv[4];
#pragma unroll
    for (int i = 0; i < 2; ++i)
      af[i] = frag_from_row(&As[buf][wr * 32 + i * 16 + lm][0], lh);
#pragma unroll
    for (int j = 0; j < 4; ++j)
      bfv[j] = frag_from_row(&Bs[buf][wc * 64 + j * 16 + lm][0], lh);
#pragma unroll
    for (int i = 0; i < 2; ++i)
#pragma unroll
      for (int j = 0; j < 4; ++j)
        acc[i][j] = __builtin_amdgcn_wmma_f32_16x16x32_bf16(
            false, af[i], false, bfv[j], (short)0, acc[i][j], false, false);
    if (more) {
#pragma unroll
      for (int j = 0; j < 16; ++j) Bs[nbuf][bn + j][bk] = breg[j];
    }
    wait_async0();
    __syncthreads();
  }
#pragma unroll
  for (int i = 0; i < 2; ++i) {
#pragma unroll
    for (int j = 0; j < 4; ++j) {
      const int n = n0 + wc * 64 + j * 16 + lm;
      const float bv = bias[n];
#pragma unroll
      for (int r = 0; r < 8; ++r) {
        const int m = m0 + wr * 32 + i * 16 + lh * 8 + r;
        C[(size_t)m * N + n] = (OUT)(acc[i][j][r] + bv);
      }
    }
  }
}

// ---------------------------------------------------------------------------
// Gram: Aout[b,t,s] = eta * (sum_h X[b,t,h]*X[b,s,h] + 1)
// B operand is X^T -> "transposed" tile is just contiguous X rows, so both
// tiles use async-to-LDS DMA, double-buffered.
// ---------------------------------------------------------------------------
__global__ __launch_bounds__(256) void gram_kernel(
    const __bf16* __restrict__ Xb, float* __restrict__ Aout)
{
  __shared__ __align__(16) __bf16 As[2][BM][LDP];
  __shared__ __align__(16) __bf16 Bs[2][BN][LDP];
  const int b = blockIdx.z;
  const __bf16* X = Xb + (size_t)b * S_ * H_;
  const int tid = threadIdx.x;
  const int m0 = blockIdx.y * BM, n0 = blockIdx.x * BN;
  const int w = tid >> 5, lane = tid & 31;
  const int wr = w >> 1, wc = w & 1;
  const int lm = lane & 15, lh = lane >> 4;

  v8f acc[2][4];
  const v8f vzero = {0.f, 0.f, 0.f, 0.f, 0.f, 0.f, 0.f, 0.f};
#pragma unroll
  for (int i = 0; i < 2; ++i)
#pragma unroll
    for (int j = 0; j < 4; ++j) acc[i][j] = vzero;

  const int ar = tid >> 1, ac = (tid & 1) << 4;

  copy32_g2l(X + (size_t)(m0 + ar) * H_ + ac, &As[0][ar][ac]);
  copy32_g2l(X + (size_t)(n0 + ar) * H_ + ac, &Bs[0][ar][ac]);
  wait_async0();
  __syncthreads();

  int buf = 0;
  for (int k0 = 0; k0 < H_; k0 += BK, buf ^= 1) {
    const int nbuf = buf ^ 1;
    if (k0 + BK < H_) {
      copy32_g2l(X + (size_t)(m0 + ar) * H_ + k0 + BK + ac, &As[nbuf][ar][ac]);
      copy32_g2l(X + (size_t)(n0 + ar) * H_ + k0 + BK + ac, &Bs[nbuf][ar][ac]);
    }
    v16bf af[2], bfv[4];
#pragma unroll
    for (int i = 0; i < 2; ++i)
      af[i] = frag_from_row(&As[buf][wr * 32 + i * 16 + lm][0], lh);
#pragma unroll
    for (int j = 0; j < 4; ++j)
      bfv[j] = frag_from_row(&Bs[buf][wc * 64 + j * 16 + lm][0], lh);
#pragma unroll
    for (int i = 0; i < 2; ++i)
#pragma unroll
      for (int j = 0; j < 4; ++j)
        acc[i][j] = __builtin_amdgcn_wmma_f32_16x16x32_bf16(
            false, af[i], false, bfv[j], (short)0, acc[i][j], false, false);
    wait_async0();
    __syncthreads();
  }
  float* Ab = Aout + (size_t)b * S_ * S_;
#pragma unroll
  for (int i = 0; i < 2; ++i)
#pragma unroll
    for (int j = 0; j < 4; ++j) {
      const int n = n0 + wc * 64 + j * 16 + lm;
#pragma unroll
      for (int r = 0; r < 8; ++r) {
        const int m = m0 + wr * 32 + i * 16 + lh * 8 + r;
        Ab[(size_t)m * S_ + n] = ETA * (acc[i][j][r] + 1.0f);
      }
    }
}

// ---------------------------------------------------------------------------
// Triangular solve + fused silu(inner) epilogue.
// One WG per (batch, 64-column tile of I). Error history lives in LDS (bf16).
// Chunks of 16 rows: WMMA for the s<t0 part, 16-step scalar recurrence inside.
// T is target (bf16) on input, overwritten in place with bf16(silu(err+T)).
// ---------------------------------------------------------------------------
#define TI 64
#define CT 16

__global__ __launch_bounds__(128) void solve_kernel(
    const float* __restrict__ U, const float* __restrict__ Amat,
    __bf16* __restrict__ T)
{
  __shared__ __bf16 ErrS[TI][S_ + 4];      // [i_local][s]  (bf16 GEMM operand)
  __shared__ float  Ast[CT][S_ + 4];       // A rows of current chunk, prefix
  __shared__ float  Ech[CT][TI + 4];       // chunk error tile (fp32)
  const int b  = blockIdx.y;
  const int i0 = blockIdx.x * TI;
  const int tid = threadIdx.x;
  const int w = tid >> 5, lane = tid & 31;
  const int lm = lane & 15, lh = lane >> 4;
  const float* Ub = U + (size_t)b * S_ * I_;
  const float* Ab = Amat + (size_t)b * S_ * S_;
  __bf16* Tb = T + (size_t)b * S_ * I_;
  const v8f vzero = {0.f, 0.f, 0.f, 0.f, 0.f, 0.f, 0.f, 0.f};

  for (int t0 = 0; t0 < S_; t0 += CT) {
    const int len = t0 + CT;
    for (int idx = tid; idx < CT * len; idx += 128) {
      const int r = idx / len, s = idx - r * len;
      Ast[r][s] = Ab[(size_t)(t0 + r) * S_ + s];
    }
    __syncthreads();

    // acc[r] = sum_{s < t0} A[t0+row, s] * Err[s, i]   (wave w -> cols 16w..)
    v8f acc = vzero;
    for (int k0 = 0; k0 < t0; k0 += 32) {
      const v16bf af  = frag_from_row_f32(&Ast[lm][k0], lh);
      const v16bf bfv = frag_from_row(&ErrS[w * 16 + lm][k0], lh);
      acc = __builtin_amdgcn_wmma_f32_16x16x32_bf16(
          false, af, false, bfv, (short)0, acc, false, false);
    }
#pragma unroll
    for (int r = 0; r < 8; ++r) {
      const int rr = lh * 8 + r;
      const int cc = w * 16 + lm;
      Ech[rr][cc] = Ub[(size_t)(t0 + rr) * I_ + i0 + cc] - acc[r];
    }
    __syncthreads();

    if (tid < TI) {
      const int c = tid;
      float e[CT];
#pragma unroll
      for (int r = 0; r < CT; ++r) e[r] = Ech[r][c];
      for (int r = 1; r < CT; ++r) {
        float a2 = 0.f;
        for (int s = 0; s < r; ++s) a2 += Ast[r][t0 + s] * e[s];
        e[r] -= a2;
      }
#pragma unroll
      for (int r = 0; r < CT; ++r) {
        ErrS[c][t0 + r] = (__bf16)e[r];
        const size_t gi = (size_t)(t0 + r) * I_ + i0 + c;
        const float inner = e[r] + (float)Tb[gi];
        const float sg = 1.0f / (1.0f + __expf(-inner));
        Tb[gi] = (__bf16)(inner * sg);       // silu(inner), in place
      }
    }
    __syncthreads();
  }
}

// ---------------------------------------------------------------------------
// Gate GEMM: Gpre = Xb@Wgx + bf16(TTT)@Wgt + bg  (K = 2*H in two phases)
// ---------------------------------------------------------------------------
__global__ __launch_bounds__(256) void gate_kernel(
    const __bf16* __restrict__ Xb, const float* __restrict__ TTT,
    const __bf16* __restrict__ Wgx, const __bf16* __restrict__ Wgt,
    const float* __restrict__ bg, __bf16* __restrict__ Gpre)
{
  __shared__ __align__(16) __bf16 As[BM][LDP];
  __shared__ __align__(16) __bf16 Bs[BN][LDP];
  const int tid = threadIdx.x;
  const int m0 = blockIdx.y * BM, n0 = blockIdx.x * BN;
  const int w = tid >> 5, lane = tid & 31;
  const int wr = w >> 1, wc = w & 1;
  const int lm = lane & 15, lh = lane >> 4;

  v8f acc[2][4];
  const v8f vzero = {0.f, 0.f, 0.f, 0.f, 0.f, 0.f, 0.f, 0.f};
#pragma unroll
  for (int i = 0; i < 2; ++i)
#pragma unroll
    for (int j = 0; j < 4; ++j) acc[i][j] = vzero;

  const int ar = tid >> 1, ac = (tid & 1) << 4;
  const int bk = tid >> 3, bn = (tid & 7) << 4;

  for (int k0 = 0; k0 < 2 * H_; k0 += BK) {
    const bool ph = (k0 >= H_);
    const int kk = ph ? k0 - H_ : k0;
    if (!ph) {
      const __bf16* asrc = Xb + (size_t)(m0 + ar) * H_ + kk + ac;
#pragma unroll
      for (int j = 0; j < 16; ++j) As[ar][ac + j] = asrc[j];
    } else {
      const float* asrc = TTT + (size_t)(m0 + ar) * H_ + kk + ac;
#pragma unroll
      for (int j = 0; j < 16; ++j) As[ar][ac + j] = (__bf16)asrc[j];
    }
    const __bf16* Wsrc = ph ? Wgt : Wgx;
    const __bf16* bsrc = Wsrc + (size_t)(kk + bk) * H_ + n0 + bn;
#pragma unroll
    for (int j = 0; j < 16; ++j) Bs[bn + j][bk] = bsrc[j];
    if (k0 + BK < 2 * H_) {
      __builtin_prefetch(bsrc + (size_t)BK * H_, 0, 0);
    }
    __syncthreads();
    v16bf af[2], bfv[4];
#pragma unroll
    for (int i = 0; i < 2; ++i)
      af[i] = frag_from_row(&As[wr * 32 + i * 16 + lm][0], lh);
#pragma unroll
    for (int j = 0; j < 4; ++j)
      bfv[j] = frag_from_row(&Bs[wc * 64 + j * 16 + lm][0], lh);
#pragma unroll
    for (int i = 0; i < 2; ++i)
#pragma unroll
      for (int j = 0; j < 4; ++j)
        acc[i][j] = __builtin_amdgcn_wmma_f32_16x16x32_bf16(
            false, af[i], false, bfv[j], (short)0, acc[i][j], false, false);
    __syncthreads();
  }
#pragma unroll
  for (int i = 0; i < 2; ++i)
#pragma unroll
    for (int j = 0; j < 4; ++j) {
      const int n = n0 + wc * 64 + j * 16 + lm;
      const float bv = bg[n];
#pragma unroll
      for (int r = 0; r < 8; ++r) {
        const int m = m0 + wr * 32 + i * 16 + lh * 8 + r;
        Gpre[(size_t)m * H_ + n] = (__bf16)(acc[i][j][r] + bv);
      }
    }
}

// ---------------------------------------------------------------------------
// Final: g = sigmoid(Gpre); y = g*ttt + (1-g)*x; layernorm over H -> out.
// One block per (b,s) row.
// ---------------------------------------------------------------------------
__global__ __launch_bounds__(256) void final_kernel(
    const __bf16* __restrict__ Gpre, const float* __restrict__ TTT,
    const float* __restrict__ x, const float* __restrict__ gamma,
    const float* __restrict__ beta, float* __restrict__ out)
{
  __shared__ float r1[256], r2[256];
  const int row = blockIdx.x;
  const size_t base = (size_t)row * H_;
  const int tid = threadIdx.x;
  float y[4];
  float s1 = 0.f, s2 = 0.f;
#pragma unroll
  for (int j = 0; j < 4; ++j) {
    const int h = tid + j * 256;
    const float g = 1.0f / (1.0f + __expf(-(float)Gpre[base + h]));
    const float t = TTT[base + h];
    const float xv = x[base + h];
    y[j] = g * t + (1.0f - g) * xv;
    s1 += y[j];
    s2 += y[j] * y[j];
  }
  r1[tid] = s1; r2[tid] = s2;
  __syncthreads();
  for (int off = 128; off > 0; off >>= 1) {
    if (tid < off) { r1[tid] += r1[tid + off]; r2[tid] += r2[tid + off]; }
    __syncthreads();
  }
  const float mu  = r1[0] * (1.0f / H_);
  const float var = r2[0] * (1.0f / H_) - mu * mu;
  const float rs  = rsqrtf(var + 1e-5f);
#pragma unroll
  for (int j = 0; j < 4; ++j) {
    const int h = tid + j * 256;
    out[base + h] = (y[j] - mu) * rs * gamma[h] + beta[h];
  }
}

// ---------------------------------------------------------------------------
extern "C" void kernel_launch(void* const* d_in, const int* in_sizes, int n_in,
                              void* d_out, int out_size, void* d_ws,
                              size_t ws_size, hipStream_t stream)
{
  (void)in_sizes; (void)n_in; (void)out_size; (void)ws_size;
  const float* x      = (const float*)d_in[0];
  const float* W_init = (const float*)d_in[1];
  const float* b_init = (const float*)d_in[2];
  const float* Wt     = (const float*)d_in[3];
  const float* bt     = (const float*)d_in[4];
  const float* Wo     = (const float*)d_in[5];
  const float* bo     = (const float*)d_in[6];
  const float* Wg     = (const float*)d_in[7];
  const float* bg     = (const float*)d_in[8];
  const float* gamma  = (const float*)d_in[9];
  const float* beta   = (const float*)d_in[10];
  float* out = (float*)d_out;

  const size_t nBSH = (size_t)B_ * S_ * H_;
  char* p = (char*)d_ws;
  __bf16* Xb   = (__bf16*)p; p += nBSH * 2;                  //  64 MB
  float*  Ubuf = (float*)p;  p += nBSH * 4;                  // 128 MB (U, later TTT)
  float*  Abuf = (float*)p;  p += (size_t)B_ * S_ * S_ * 4;  //  64 MB (A, later Gpre)
  __bf16* Tbuf = (__bf16*)p; p += nBSH * 2;                  //  64 MB (target -> silu(inner))
  __bf16* Wd   = (__bf16*)p; p += (size_t)H_ * I_ * 2;
  __bf16* Wtb  = (__bf16*)p; p += (size_t)H_ * I_ * 2;
  __bf16* Wob  = (__bf16*)p; p += (size_t)I_ * H_ * 2;
  __bf16* Wgx  = (__bf16*)p; p += (size_t)H_ * H_ * 2;
  __bf16* Wgt  = (__bf16*)p; p += (size_t)H_ * H_ * 2;
  float*  bias_u = (float*)p; p += (size_t)I_ * 4;
  __bf16* Gpre = (__bf16*)Abuf;  // reuse after solve consumed A
  float*  TTT  = Ubuf;           // reuse after solve consumed U

  pack_kernel<<<2048, 256, 0, stream>>>(x, W_init, b_init, Wt, bt, Wo, Wg,
                                        Xb, Wd, Wtb, Wob, Wgx, Wgt, bias_u);

  const dim3 gg(I_ / BN, (B_ * S_) / BM);  // (8, 256)
  // U = X@(W0 - Wt) + (b0 - bt)
  gemm_bf16_kernel<float><<<gg, 256, 0, stream>>>(Xb, Wd, bias_u, Ubuf,
                                                  B_ * S_, I_, H_);
  // target = X@Wt + bt  (bf16)
  gemm_bf16_kernel<__bf16><<<gg, 256, 0, stream>>>(Xb, Wtb, bt, Tbuf,
                                                   B_ * S_, I_, H_);
  // A = eta*(X X^T + 1) per batch
  gram_kernel<<<dim3(S_ / BN, S_ / BM, B_), 256, 0, stream>>>(Xb, Abuf);
  // triangular solve + silu(inner) fused (Tbuf in-place)
  solve_kernel<<<dim3(I_ / TI, B_), 128, 0, stream>>>(Ubuf, Abuf, Tbuf);
  // ttt = silu(inner)@Wo + bo  (fp32, into Ubuf)
  gemm_bf16_kernel<float><<<gg, 256, 0, stream>>>(Tbuf, Wob, bo, TTT,
                                                  B_ * S_, H_, I_);
  // gate pre-activation (bf16, into Abuf)
  gate_kernel<<<gg, 256, 0, stream>>>(Xb, TTT, Wgx, Wgt, bg, Gpre);
  // combine + layernorm
  final_kernel<<<B_ * S_, 256, 0, stream>>>(Gpre, TTT, x, gamma, beta, out);
}
